// ContactCNN_2147483648696
// MI455X (gfx1250) — matrix operand order repaired and checked by reading, not compiled, not run on hardware
//
#include <hip/hip_runtime.h>
#include <hip/hip_bf16.h>

typedef __attribute__((ext_vector_type(16))) _Float16 v16h;
typedef __attribute__((ext_vector_type(8)))  _Float16 v8h;
typedef __attribute__((ext_vector_type(8)))  float    v8f;

#define LL   512
#define DD   128
#define HH   50
#define HP   64
#define BN_EPS 1e-5f

// ---------------------------------------------------------------------------
// Pass 0: zero the stats region of the workspace (must happen every launch)
// ---------------------------------------------------------------------------
__global__ void zero_stats_k(float* stats) {
    int t = blockIdx.x * 256 + threadIdx.x;
    if (t < 512) stats[t] = 0.f;
}

// |a-b| without fabs: avoids a gfx1250 ISel crash on vectorized fabs bitcasts.
__device__ __forceinline__ float absdiff(float a, float b) {
    return fmaxf(a - b, b - a);
}

// ---------------------------------------------------------------------------
// Pass 1: WMMA pairwise GEMM.
// One wave = one (b, i, j-tile of 16). D[16 j's x 64 ch] = A(16x256 feats) x B(256x64 w)
// Features: k<128 -> |x1[i,k]-x2[j,k]| ; k>=128 -> x1[i,k-128]*x2[j,k-128]
// conv1_w rows are [w_dif(128) | w_mul(128)] which matches the feature ordering.
// ---------------------------------------------------------------------------
__global__ __launch_bounds__(256)
void pass1_wmma(const float* __restrict__ x1, const float* __restrict__ x2,
                const float* __restrict__ w1, const float* __restrict__ b1,
                _Float16* __restrict__ hws)
{
    __shared__ _Float16 wB[HP * 256];          // 32 KB: f16 conv1_w, zero-padded to 64 ch

    const int tid = threadIdx.x;
    for (int idx = tid; idx < HP * 256; idx += 256) {
        int ch = idx >> 8, k = idx & 255;
        wB[idx] = (ch < HH) ? (_Float16)w1[ch * 256 + k] : (_Float16)0.f;
    }
    __syncthreads();

    const int wave = tid >> 5;
    const int lane = tid & 31;
    const int half = lane >> 4;                // K-chunk selector within fragments
    const int m    = lane & 15;                // A row (j offset) / B column (channel)

    const int job = blockIdx.x * 8 + wave;     // 65536 jobs total
    const int jt  = job & 31;
    const int i   = (job >> 5) & 511;
    const int b   = job >> 14;

    const float* x1r = x1 + ((size_t)(b * LL + i)) * DD;
    const float* x2r = x2 + ((size_t)(b * LL + jt * 16 + m)) * DD;
    __builtin_prefetch(x1r, 0, 1);             // global_prefetch_b8
    __builtin_prefetch(x2r, 0, 1);

    // Accumulators seeded with conv1 bias (per D layout: N = channel = lane&15)
    v8f acc[4];
#pragma unroll
    for (int t = 0; t < 4; ++t) {
        int ch = t * 16 + m;
        float bb = (ch < HH) ? b1[ch] : 0.f;
        acc[t] = (v8f){bb, bb, bb, bb, bb, bb, bb, bb};
    }

#pragma unroll
    for (int ks = 0; ks < 8; ++ks) {
        const int kb = ks * 32;
        v16h A;
        // A 16-bit 16x32 layout: lane<16 K 0..15, lane>=16 K 16..31 (contiguous, matches B)
#pragma unroll
        for (int g = 0; g < 2; ++g) {
            const int k0  = kb + half * 16 + g * 8;       // kb decides abs vs mul (uniform per ks)
            const bool ismul = (kb >= 128);
            const int off = ismul ? (k0 - 128) : k0;
            const float4* p1 = (const float4*)(x1r + off);
            const float4* p2 = (const float4*)(x2r + off);
#pragma unroll
            for (int q = 0; q < 2; ++q) {
                float4 u = p1[q], v = p2[q];
                float r0, r1, r2, r3;
                if (ismul) { r0 = u.x * v.x; r1 = u.y * v.y; r2 = u.z * v.z; r3 = u.w * v.w; }
                else { r0 = absdiff(u.x, v.x); r1 = absdiff(u.y, v.y);
                       r2 = absdiff(u.z, v.z); r3 = absdiff(u.w, v.w); }
                const int hb = g * 8 + q * 4;
                A[hb + 0] = (_Float16)r0; A[hb + 1] = (_Float16)r1;
                A[hb + 2] = (_Float16)r2; A[hb + 3] = (_Float16)r3;
            }
        }
        // B fragments: lane n = column (channel), contiguous K run of 16
        const _Float16* bk = wB + kb + half * 16;
#pragma unroll
        for (int t = 0; t < 4; ++t) {
            const int ch = t * 16 + m;
            v16h B = *(const v16h*)(bk + ch * 256);
            acc[t] = __builtin_amdgcn_wmma_f32_16x16x32_f16(
                false, A, false, B, (short)0, acc[t], false, false);
        }
    }

    // Store: relu + f16 pack. D layout: lane<16 -> j = jt*16+v ; lane>=16 -> jt*16+8+v
    const int j0 = jt * 16 + half * 8;
#pragma unroll
    for (int t = 0; t < 4; ++t) {
        const int ch = t * 16 + m;
        if (ch < HH) {
            v8h o;
#pragma unroll
            for (int v = 0; v < 8; ++v) {
                float x = acc[t][v];
                o[v] = (_Float16)(x > 0.f ? x : 0.f);
            }
            *(v8h*)(hws + ((((size_t)b * HH + ch) * LL + i) * LL + j0)) = o;
        }
    }
}

// ---------------------------------------------------------------------------
// Pass 2: per-channel sum / sumsq over (b, i, j) for BN1
// ---------------------------------------------------------------------------
__global__ __launch_bounds__(256)
void bn1_stats_k(const _Float16* __restrict__ hws, float* __restrict__ stats) {
    const int ch = blockIdx.x, b = blockIdx.y, zc = blockIdx.z;   // 50 x 4 x 16
    const _Float16* p = hws + (((size_t)b * HH + ch) * LL + zc * 32) * LL;
    float s = 0.f, q = 0.f;
    for (int t = threadIdx.x; t < 32 * LL; t += 256) {
        float v = (float)p[t];
        s += v; q += v * v;
    }
    __shared__ float rs[256], rq[256];
    rs[threadIdx.x] = s; rq[threadIdx.x] = q;
    __syncthreads();
    for (int w = 128; w > 0; w >>= 1) {
        if (threadIdx.x < w) { rs[threadIdx.x] += rs[threadIdx.x + w];
                               rq[threadIdx.x] += rq[threadIdx.x + w]; }
        __syncthreads();
    }
    if (threadIdx.x == 0) {
        atomicAdd(&stats[ch], rs[0]);
        atomicAdd(&stats[64 + ch], rq[0]);
    }
}

__global__ void bn1_fin_k(float* stats, const float* g, const float* be) {
    int ch = threadIdx.x;
    if (ch < HH) {
        float n = (float)(4 * LL * LL);
        float mean = stats[ch] / n;
        float var  = stats[64 + ch] / n - mean * mean;
        float a = g[ch] * rsqrtf(var + BN_EPS);
        stats[128 + ch] = a;                 // scale
        stats[192 + ch] = be[ch] - mean * a; // shift
    }
}

// ---------------------------------------------------------------------------
// Pass 3: 7x7 SAME conv (50 -> 1 ch), BN1 applied at LDS-tile load (padding zeros
// are post-BN, matching reference). 32x32 output tile per block; each thread
// computes 4 outputs along j with a 10-wide register window (2.8x fewer LDS reads).
// Also accumulates BN2 stats.
// ---------------------------------------------------------------------------
__global__ __launch_bounds__(256)
void conv2_k(const _Float16* __restrict__ hws, const float* __restrict__ w2,
             const float* __restrict__ c2b, float* __restrict__ stats,
             float* __restrict__ yws)
{
    __shared__ float tile[38][40];
    __shared__ float wl[HH * 49];
    __shared__ float rs[256], rq[256];

    const int tid = threadIdx.x;
    for (int idx = tid; idx < HH * 49; idx += 256) wl[idx] = w2[idx];

    const int jt = blockIdx.x, it = blockIdx.y, b = blockIdx.z;   // 16 x 16 x 4
    const int ti  = tid >> 3;            // 0..31 : output row within tile
    const int tj  = (tid & 7) * 4;       // 0..28 : output col base within tile
    const int oi  = it * 32 + ti;
    const int ojb = jt * 32 + tj;

    const float bias = c2b[0];
    float acc[4] = {bias, bias, bias, bias};

    for (int ch = 0; ch < HH; ++ch) {
        const float a1 = stats[128 + ch], c1 = stats[192 + ch];
        __syncthreads();
        for (int idx = tid; idx < 38 * 38; idx += 256) {
            int r = idx / 38, c = idx % 38;
            int gi = it * 32 + r - 3, gj = jt * 32 + c - 3;
            float v = 0.f;
            if (gi >= 0 && gi < LL && gj >= 0 && gj < LL)
                v = (float)hws[(((size_t)b * HH + ch) * LL + gi) * LL + gj] * a1 + c1;
            tile[r][c] = v;
        }
        __syncthreads();
#pragma unroll
        for (int di = 0; di < 7; ++di) {
            float row[10];
#pragma unroll
            for (int c = 0; c < 10; ++c) row[c] = tile[ti + di][tj + c];
#pragma unroll
            for (int dj = 0; dj < 7; ++dj) {
                const float w = wl[ch * 49 + di * 7 + dj];
#pragma unroll
                for (int p = 0; p < 4; ++p) acc[p] += row[dj + p] * w;
            }
        }
    }

    float4 o; o.x = acc[0]; o.y = acc[1]; o.z = acc[2]; o.w = acc[3];
    *(float4*)(yws + ((size_t)b * LL + oi) * LL + ojb) = o;

    __syncthreads();
    rs[tid] = acc[0] + acc[1] + acc[2] + acc[3];
    rq[tid] = acc[0]*acc[0] + acc[1]*acc[1] + acc[2]*acc[2] + acc[3]*acc[3];
    __syncthreads();
    for (int w = 128; w > 0; w >>= 1) {
        if (tid < w) { rs[tid] += rs[tid + w]; rq[tid] += rq[tid + w]; }
        __syncthreads();
    }
    if (tid == 0) {
        atomicAdd(&stats[256], rs[0]);
        atomicAdd(&stats[257], rq[0]);
    }
}

__global__ void bn2_fin_k(float* stats, const float* g2, const float* b2) {
    if (threadIdx.x == 0) {
        float n = (float)(4 * LL * LL);
        float mean = stats[256] / n;
        float var  = stats[257] / n - mean * mean;
        float a = g2[0] * rsqrtf(var + BN_EPS);
        stats[258] = a;
        stats[259] = b2[0] - mean * a;
    }
}

__global__ void final_k(const float* __restrict__ yws, const float* __restrict__ stats,
                        float* __restrict__ out) {
    int idx = blockIdx.x * 256 + threadIdx.x;
    float z = yws[idx] * stats[258] + stats[259];
    out[idx] = 1.f / (1.f + __expf(-z));
}

// ---------------------------------------------------------------------------
extern "C" void kernel_launch(void* const* d_in, const int* in_sizes, int n_in,
                              void* d_out, int out_size, void* d_ws, size_t ws_size,
                              hipStream_t stream) {
    const float* x1  = (const float*)d_in[0];
    const float* x2  = (const float*)d_in[1];
    const float* w1  = (const float*)d_in[2];
    const float* b1  = (const float*)d_in[3];
    const float* g1  = (const float*)d_in[4];
    const float* be1 = (const float*)d_in[5];
    const float* w2  = (const float*)d_in[6];
    const float* b2  = (const float*)d_in[7];
    const float* g2  = (const float*)d_in[8];
    const float* be2 = (const float*)d_in[9];
    (void)be2; (void)in_sizes; (void)n_in; (void)ws_size; (void)out_size;

    // workspace layout: [0,2048) stats ; then h (f16, 4*50*512*512) ; then y (f32)
    float*     stats = (float*)d_ws;
    _Float16*  hws   = (_Float16*)((char*)d_ws + 2048);
    float*     yws   = (float*)((char*)d_ws + 2048 + (size_t)4 * HH * LL * LL * 2);
    float*     out   = (float*)d_out;

    zero_stats_k<<<dim3(2), dim3(256), 0, stream>>>(stats);
    pass1_wmma<<<dim3(8192), dim3(256), 0, stream>>>(x1, x2, w1, b1, hws);
    bn1_stats_k<<<dim3(50, 4, 16), dim3(256), 0, stream>>>(hws, stats);
    bn1_fin_k<<<dim3(1), dim3(64), 0, stream>>>(stats, g1, be1);
    conv2_k<<<dim3(16, 16, 4), dim3(256), 0, stream>>>(hws, w2, b2, stats, yws);
    bn2_fin_k<<<dim3(1), dim3(1), 0, stream>>>(stats, g2, be2);
    final_k<<<dim3(4096), dim3(256), 0, stream>>>(yws, stats, out);
}
